// Heat2D_40956808135070
// MI455X (gfx1250) — matrix operand order repaired and checked by reading, not compiled, not run on hardware
//
#include <hip/hip_runtime.h>
#include <hip/hip_bf16.h>
#include <math.h>

#ifndef M_PI
#define M_PI 3.14159265358979323846
#endif

#define BB 32
#define CC 192
#define HH 64
#define WWX 64
#define HWX (HH*WWX)      // 4096
#define TOKN (BB*HWX)     // 131072
#define C2 (2*CC)         // 384

typedef __bf16 bf16_t;
typedef __attribute__((ext_vector_type(16))) __bf16 v16bf;
typedef __attribute__((ext_vector_type(8)))  float  v8f;

static __device__ __forceinline__ bf16_t tobf(float f) { return (bf16_t)f; }
static __device__ __forceinline__ float  fbf(bf16_t b) { return (float)b; }

static __device__ __forceinline__ v8f vzero8() {
  v8f z;
#pragma unroll
  for (int i = 0; i < 8; ++i) z[i] = 0.f;
  return z;
}

// ---- WMMA fragment helpers (wave32, 16x16x32 bf16, contiguous k order k = 16*(lane/16) + e for A and B) ----

// A-matrix 16x32: lane<16 -> m=lane, K={0..15}; lane>=16 -> m=lane-16, K={16..31}
static __device__ __forceinline__ v16bf frag_a(const bf16_t* s, int row0, int k0, int ld) {
  int lane = threadIdx.x & 31;
  const bf16_t* p = s + (size_t)(row0 + (lane & 15)) * ld + k0 + ((lane >> 4) << 4);
  v16bf a;
#pragma unroll
  for (int e = 0; e < 16; ++e) a[e] = p[e];
  return a;
}
// B-matrix 32x16 from LDS stored [n][k] (row n contiguous over k)
static __device__ __forceinline__ v16bf frag_b_nk(const bf16_t* s, int col0, int k0, int ld) {
  int lane = threadIdx.x & 31;
  const bf16_t* p = s + (size_t)(col0 + (lane & 15)) * ld + k0 + ((lane >> 4) << 4);
  v16bf b;
#pragma unroll
  for (int e = 0; e < 16; ++e) b[e] = p[e];
  return b;
}
// B-matrix 32x16 from LDS stored [k][n]
static __device__ __forceinline__ v16bf frag_b_kn(const bf16_t* s, int k0, int col0, int ld) {
  int lane = threadIdx.x & 31;
  const bf16_t* p = s + (size_t)(k0 + ((lane >> 4) << 4)) * ld + col0 + (lane & 15);
  v16bf b;
#pragma unroll
  for (int e = 0; e < 16; ++e) b[e] = p[e * ld];
  return b;
}
static __device__ __forceinline__ v8f wmma_bf(v16bf a, v16bf b, v8f c) {
  return __builtin_amdgcn_wmma_f32_16x16x32_bf16(false, a, false, b, (short)0, c, false, false);
}

// 64x32 bf16 tile (LDS row stride 64) global->LDS via CDNA5 async copy (ASYNCcnt path, no VGPR round-trip).
// Each of 256 threads moves one 16B chunk. Low 32 bits of the generic shared pointer
// are the LDS byte offset (aperture rule), which is what VDST of the async op wants.
static __device__ __forceinline__ void tile64_g2s(bf16_t* s, const bf16_t* g, int ldg) {
  int t = threadIdx.x;
  int r = t >> 2, c = (t & 3) << 3;
  unsigned lds = (unsigned)(size_t)(s + r * 64 + c);
  unsigned long long ga = (unsigned long long)(g + (size_t)r * ldg + c);
  asm volatile("global_load_async_to_lds_b128 %0, %1, off"
               :: "v"(lds), "v"(ga) : "memory");
}
static __device__ __forceinline__ void wait_async_all() {
  asm volatile("s_wait_asynccnt 0" ::: "memory");
}

// ================= depthwise 3x3 conv + NCHW->token(bf16) =================
__global__ void __launch_bounds__(256)
k_dwconv(const float* __restrict__ x, const float* __restrict__ w9,
         const float* __restrict__ bia, bf16_t* __restrict__ o)
{
  size_t idx = (size_t)blockIdx.x * 256 + threadIdx.x;
  if (idx >= (size_t)TOKN * CC) return;
  int w = (int)(idx & 63);
  int h = (int)((idx >> 6) & 63);
  int c = (int)((idx >> 12) % CC);
  int b = (int)(idx / ((size_t)CC * HWX));
  const float* xp = x + (((size_t)b * CC + c) << 12);
  const float* wp = w9 + c * 9;
  float s = bia[c];
#pragma unroll
  for (int dh = -1; dh <= 1; ++dh) {
#pragma unroll
    for (int dw = -1; dw <= 1; ++dw) {
      int hh = h + dh, ww = w + dw;
      if (hh >= 0 && hh < HH && ww >= 0 && ww < WWX)
        s += xp[(hh << 6) + ww] * wp[(dh + 1) * 3 + (dw + 1)];
    }
  }
  o[(size_t)((b << 12) + (h << 6) + w) * CC + c] = tobf(s);
}

// ================= token GEMM:  Y = X(Mx192) @ Wrow^T + bias =================
// mode 0: split -> xh(bf16) / silu(z)(bf16);  mode 2: write NCHW fp32 (d_out)
// mode 3: kfull = exp(-(wn^2+wm^2)*relu(v))  (M=4096)
__global__ void __launch_bounds__(256)
k_tok_gemm(const bf16_t* __restrict__ X, const bf16_t* __restrict__ Wm,
           const float* __restrict__ bias, int mode,
           bf16_t* __restrict__ o_xh, bf16_t* __restrict__ o_z,
           float* __restrict__ o_f)
{
  __shared__ __align__(16) bf16_t As[64 * 64];
  __shared__ __align__(16) bf16_t Bs[64 * 64];
  int m0 = blockIdx.x << 6, n0 = blockIdx.y << 6;
  int wid = threadIdx.x >> 5;
  int wm = (wid >> 1) << 4;
  int wn = (wid & 1) << 5;
  v8f acc0 = vzero8(), acc1 = vzero8();
  for (int kc = 0; kc < CC; kc += 64) {
    if (kc + 64 < CC)
      __builtin_prefetch(X + (size_t)(m0 + (threadIdx.x & 63)) * CC + kc + 64, 0, 1);
    __syncthreads();
    tile64_g2s(As, X + (size_t)m0 * CC + kc, CC);
    tile64_g2s(Bs, Wm + (size_t)n0 * CC + kc, CC);
    wait_async_all();
    __syncthreads();
#pragma unroll
    for (int ks = 0; ks < 2; ++ks) {
      v16bf a  = frag_a(As, wm, ks * 32, 64);
      v16bf b0 = frag_b_nk(Bs, wn,      ks * 32, 64);
      v16bf b1 = frag_b_nk(Bs, wn + 16, ks * 32, 64);
      acc0 = wmma_bf(a, b0, acc0);
      acc1 = wmma_bf(a, b1, acc1);
    }
  }
  int lane = threadIdx.x & 31;
  int nl = lane & 15, mh = (lane >> 4) << 3;
#pragma unroll
  for (int j = 0; j < 2; ++j) {
    v8f acc = j ? acc1 : acc0;
    int n = n0 + wn + (j << 4) + nl;
#pragma unroll
    for (int r = 0; r < 8; ++r) {
      int tokid = m0 + wm + mh + r;
      float v = acc[r] + bias[n];
      if (mode == 0) {
        if (n < CC) {
          o_xh[(size_t)tokid * CC + n] = tobf(v);
        } else {
          float sz = v / (1.f + __expf(-v));     // silu
          o_z[(size_t)tokid * CC + (n - CC)] = tobf(sz);
        }
      } else if (mode == 2) {
        int b = tokid >> 12, hw = tokid & 4095;
        o_f[((size_t)(b * CC + n) << 12) + hw] = v;
      } else { // mode 3
        int h = tokid >> 6, w = tokid & 63;
        float a1 = (float)M_PI * h * (1.f / 64.f);
        float a2 = (float)M_PI * w * (1.f / 64.f);
        float rv = v > 0.f ? v : 0.f;
        o_f[(size_t)tokid * CC + n] = __expf(-(a1 * a1 + a2 * a2) * rv);
      }
    }
  }
}

// ============ 64-point axis transform (real DCT or complex DFT) ============
// out[g*64+n, j] = sum_k A[n,k] * X[g*64+k, j];  complex: Re'=Ac*R+As*I, Im'=Ac*I+An*R
// outMode 0: write Rout(/Iout) bf16;  outMode 1: comb[(g*64+n)*J+j] += moe[b]*scale*Re'
__global__ void __launch_bounds__(256)
k_caxis(const bf16_t* __restrict__ Ac, const bf16_t* __restrict__ Asp,
        const bf16_t* __restrict__ Asn,
        const bf16_t* __restrict__ Rin, const bf16_t* __restrict__ Iin,
        bf16_t* __restrict__ Rout, bf16_t* __restrict__ Iout,
        float* __restrict__ comb, const float* __restrict__ moe,
        int moeIdx, float scale, int gPerB, int J, int outMode)
{
  __shared__ __align__(16) bf16_t sAc[64*64], sAs[64*64], sAn[64*64], sR[64*64], sI[64*64];
  int j0 = blockIdx.x << 6;
  int g  = blockIdx.z;
  bool cplx  = (Asp != nullptr);
  bool haveI = (Iin != nullptr);
  tile64_g2s(sAc, Ac, 64);
  if (cplx) { tile64_g2s(sAs, Asp, 64); tile64_g2s(sAn, Asn, 64); }
  tile64_g2s(sR, Rin + (size_t)g * 64 * J + j0, J);
  if (haveI) tile64_g2s(sI, Iin + (size_t)g * 64 * J + j0, J);
  wait_async_all();
  __syncthreads();
  int wid = threadIdx.x >> 5;
  int wm = (wid >> 1) << 4, wn = (wid & 1) << 5;
  v8f aR0 = vzero8(), aR1 = vzero8(), aI0 = vzero8(), aI1 = vzero8();
#pragma unroll
  for (int ks = 0; ks < 2; ++ks) {
    int k0 = ks * 32;
    v16bf ac  = frag_a(sAc, wm, k0, 64);
    v16bf bR0 = frag_b_kn(sR, k0, wn,      64);
    v16bf bR1 = frag_b_kn(sR, k0, wn + 16, 64);
    aR0 = wmma_bf(ac, bR0, aR0);
    aR1 = wmma_bf(ac, bR1, aR1);
    if (cplx) {
      v16bf an = frag_a(sAn, wm, k0, 64);
      if (haveI) {
        v16bf as  = frag_a(sAs, wm, k0, 64);
        v16bf bI0 = frag_b_kn(sI, k0, wn,      64);
        v16bf bI1 = frag_b_kn(sI, k0, wn + 16, 64);
        aR0 = wmma_bf(as, bI0, aR0);
        aR1 = wmma_bf(as, bI1, aR1);
        if (outMode == 0) { aI0 = wmma_bf(ac, bI0, aI0); aI1 = wmma_bf(ac, bI1, aI1); }
      }
      if (outMode == 0) { aI0 = wmma_bf(an, bR0, aI0); aI1 = wmma_bf(an, bR1, aI1); }
    }
  }
  int lane = threadIdx.x & 31;
  int nl = lane & 15, mh = (lane >> 4) << 3;
#pragma unroll
  for (int j = 0; j < 2; ++j) {
    v8f accR = j ? aR1 : aR0;
    v8f accI = j ? aI1 : aI0;
    int col = j0 + wn + (j << 4) + nl;
#pragma unroll
    for (int r = 0; r < 8; ++r) {
      int mrow = wm + mh + r;
      size_t oidx = ((size_t)g * 64 + mrow) * J + col;
      if (outMode == 0) {
        Rout[oidx] = tobf(accR[r]);
        if (cplx) Iout[oidx] = tobf(accI[r]);
      } else {
        int b = g / gPerB;
        comb[oidx] += moe[b * 4 + moeIdx] * scale * accR[r];
      }
    }
  }
}

// ============ 192-point complex channel DFT (token GEMM, 4 fused real GEMMs) ============
// Re' = R@C^T + I@S^T ; Im' = I@C^T + R@Sn^T   (fwd: S=+sin,Sn=-sin; inv: swapped)
__global__ void __launch_bounds__(256)
k_cchan(const bf16_t* __restrict__ Rin, const bf16_t* __restrict__ Iin,
        const bf16_t* __restrict__ Wc, const bf16_t* __restrict__ Wsp,
        const bf16_t* __restrict__ Wsn,
        bf16_t* __restrict__ Rout, bf16_t* __restrict__ Iout,
        float* __restrict__ comb, const float* __restrict__ moe,
        int moeIdx, float scale, int outMode)
{
  __shared__ __align__(16) bf16_t sR[64*64], sI[64*64], sC[64*64], sS[64*64], sN[64*64];
  int m0 = blockIdx.x << 6, n0 = blockIdx.y << 6;
  int wid = threadIdx.x >> 5;
  int wm = (wid >> 1) << 4, wn = (wid & 1) << 5;
  v8f aR0 = vzero8(), aR1 = vzero8(), aI0 = vzero8(), aI1 = vzero8();
  for (int kc = 0; kc < CC; kc += 64) {
    __syncthreads();
    tile64_g2s(sR, Rin + (size_t)m0 * CC + kc, CC);
    tile64_g2s(sI, Iin + (size_t)m0 * CC + kc, CC);
    tile64_g2s(sC, Wc  + (size_t)n0 * CC + kc, CC);
    tile64_g2s(sS, Wsp + (size_t)n0 * CC + kc, CC);
    tile64_g2s(sN, Wsn + (size_t)n0 * CC + kc, CC);
    wait_async_all();
    __syncthreads();
#pragma unroll
    for (int ks = 0; ks < 2; ++ks) {
      int k0 = ks * 32;
      v16bf aR = frag_a(sR, wm, k0, 64);
      v16bf aI = frag_a(sI, wm, k0, 64);
      v16bf c0 = frag_b_nk(sC, wn,      k0, 64);
      v16bf c1 = frag_b_nk(sC, wn + 16, k0, 64);
      v16bf s0 = frag_b_nk(sS, wn,      k0, 64);
      v16bf s1 = frag_b_nk(sS, wn + 16, k0, 64);
      aR0 = wmma_bf(aR, c0, aR0); aR0 = wmma_bf(aI, s0, aR0);
      aR1 = wmma_bf(aR, c1, aR1); aR1 = wmma_bf(aI, s1, aR1);
      if (outMode == 0) {
        v16bf n0f = frag_b_nk(sN, wn,      k0, 64);
        v16bf n1f = frag_b_nk(sN, wn + 16, k0, 64);
        aI0 = wmma_bf(aI, c0, aI0); aI0 = wmma_bf(aR, n0f, aI0);
        aI1 = wmma_bf(aI, c1, aI1); aI1 = wmma_bf(aR, n1f, aI1);
      }
    }
  }
  int lane = threadIdx.x & 31;
  int nl = lane & 15, mh = (lane >> 4) << 3;
#pragma unroll
  for (int j = 0; j < 2; ++j) {
    v8f accR = j ? aR1 : aR0;
    v8f accI = j ? aI1 : aI0;
    int n = n0 + wn + (j << 4) + nl;
#pragma unroll
    for (int r = 0; r < 8; ++r) {
      int tokid = m0 + wm + mh + r;
      size_t oidx = (size_t)tokid * CC + n;
      if (outMode == 0) {
        Rout[oidx] = tobf(accR[r]);
        Iout[oidx] = tobf(accI[r]);
      } else {
        int b = tokid >> 12;
        comb[oidx] += moe[b * 4 + moeIdx] * scale * accR[r];
      }
    }
  }
}

// ================= elementwise: multiply plane(s) by k =================
__global__ void __launch_bounds__(256)
k_kmul(bf16_t* __restrict__ A, bf16_t* __restrict__ Bp, const float* __restrict__ kf)
{
  size_t i = (size_t)blockIdx.x * 256 + threadIdx.x;
  if (i >= (size_t)TOKN * CC) return;
  size_t tok = i / CC;
  int c = (int)(i - tok * CC);
  float kv = kf[(size_t)(tok & 4095) * CC + c];
  A[i] = tobf(fbf(A[i]) * kv);
  if (Bp) Bp[i] = tobf(fbf(Bp[i]) * kv);
}

// ================= haar branch (per-token channel transform) =================
__global__ void __launch_bounds__(192)
k_haar(const bf16_t* __restrict__ xh, const float* __restrict__ kf,
       const float* __restrict__ moe, float* __restrict__ comb)
{
  __shared__ float s0[192], s1[192];
  size_t t = blockIdx.x;
  int c = threadIdx.x;
  s0[c] = fbf(xh[t * CC + c]);
  __syncthreads();
  { // fwd pass over 64-chunks
    int q = c >> 6, r = c & 63;
    float v;
    if (r < 32) v = (s0[q * 64 + 2 * r] + s0[q * 64 + 2 * r + 1]) * 0.5f;
    else { int rr = r - 32; v = (s0[q * 64 + 2 * rr] - s0[q * 64 + 2 * rr + 1]) * 0.5f; }
    __syncthreads(); s1[c] = v; __syncthreads();
  }
  { // fwd pass over full 192 row
    float v;
    if (c < 96) v = (s1[2 * c] + s1[2 * c + 1]) * 0.5f;
    else { int rr = c - 96; v = (s1[2 * rr] - s1[2 * rr + 1]) * 0.5f; }
    __syncthreads(); s0[c] = v * kf[(size_t)(t & 4095) * CC + c]; __syncthreads();
  }
  { // inv pass over 64-chunks
    int q = c >> 6, r = c & 63, j = r >> 1;
    float a = s0[q * 64 + j], d = s0[q * 64 + 32 + j];
    float v = (r & 1) ? (a - d) : (a + d);
    __syncthreads(); s1[c] = v; __syncthreads();
  }
  { // inv pass over full 192 row
    int j = c >> 1;
    float a = s1[j], d = s1[96 + j];
    float v = (c & 1) ? (a - d) : (a + d);
    int b = (int)(t >> 12);
    comb[t * CC + c] += moe[b * 4 + 2] * v;
  }
}

// ================= mean-pool xh over HW =================
__global__ void __launch_bounds__(192)
k_pool(const bf16_t* __restrict__ xh, float* __restrict__ pooled)
{
  int b = blockIdx.x, c = threadIdx.x;
  float acc = 0.f;
  const bf16_t* p = xh + ((size_t)b << 12) * CC + c;
  for (int t = 0; t < HWX; ++t) acc += fbf(p[(size_t)t * CC]);
  pooled[b * CC + c] = acc * (1.f / HWX);
}

// ================= MLP + batchnorms + gumbel argmax -> moe one-hot =================
__global__ void __launch_bounds__(256)
k_mlp(const float* __restrict__ pooled,
      const float* __restrict__ p1w, const float* __restrict__ p1b,
      const float* __restrict__ g1,  const float* __restrict__ b1,
      const float* __restrict__ p2w, const float* __restrict__ p2b,
      const float* __restrict__ g2,  const float* __restrict__ b2,
      const float* __restrict__ p3w, const float* __restrict__ p3b,
      const float* __restrict__ gum, float* __restrict__ moe)
{
  __shared__ float h1[32 * 256];
  __shared__ float h2[32 * 32];
  int t = threadIdx.x;
  for (int o = t; o < 32 * 256; o += 256) {
    int b = o >> 8, j = o & 255;
    float s = p1b[j];
    for (int c = 0; c < CC; ++c) s += pooled[b * CC + c] * p1w[j * CC + c];
    h1[o] = s > 0.f ? s : 0.1f * s;
  }
  __syncthreads();
  { // bn1 over batch, column t
    int j = t;
    float m = 0.f; for (int b = 0; b < 32; ++b) m += h1[b * 256 + j]; m *= (1.f / 32.f);
    float v = 0.f; for (int b = 0; b < 32; ++b) { float d = h1[b * 256 + j] - m; v += d * d; }
    float is = rsqrtf(v * (1.f / 32.f) + 1e-5f);
    for (int b = 0; b < 32; ++b) h1[b * 256 + j] = (h1[b * 256 + j] - m) * is * g1[j] + b1[j];
  }
  __syncthreads();
  for (int o = t; o < 32 * 32; o += 256) {
    int b = o >> 5, j = o & 31;
    float s = p2b[j];
    for (int c = 0; c < 256; ++c) s += h1[b * 256 + c] * p2w[j * 256 + c];
    h2[o] = s > 0.f ? s : 0.1f * s;
  }
  __syncthreads();
  if (t < 32) { // bn2 over batch, column t
    int j = t;
    float m = 0.f; for (int b = 0; b < 32; ++b) m += h2[b * 32 + j]; m *= (1.f / 32.f);
    float v = 0.f; for (int b = 0; b < 32; ++b) { float d = h2[b * 32 + j] - m; v += d * d; }
    float is = rsqrtf(v * (1.f / 32.f) + 1e-5f);
    for (int b = 0; b < 32; ++b) h2[b * 32 + j] = (h2[b * 32 + j] - m) * is * g2[j] + b2[j];
  }
  __syncthreads();
  if (t < 32) {
    int b = t;
    float l[3];
    for (int r = 0; r < 3; ++r) {
      float s = p3b[r];
      for (int c = 0; c < 32; ++c) s += h2[b * 32 + c] * p3w[r * 32 + c];
      l[r] = s + gum[b * 3 + r];
    }
    int am = 0;
    if (l[1] > l[am]) am = 1;
    if (l[2] > l[am]) am = 2;
    for (int r = 0; r < 4; ++r) moe[b * 4 + r] = (r < 3 && r == am) ? 1.f : 0.f;
  }
}

// ================= layernorm over C + silu(z) gate -> bf16 tokens =================
__global__ void __launch_bounds__(256)
k_ln_gate(const float* __restrict__ comb, const bf16_t* __restrict__ zs,
          const float* __restrict__ lng, const float* __restrict__ lnb,
          bf16_t* __restrict__ gated)
{
  __shared__ float red[256];
  size_t t = blockIdx.x;
  int c = threadIdx.x;
  float v = (c < CC) ? comb[t * CC + c] : 0.f;
  red[c] = v; __syncthreads();
  for (int s = 128; s > 0; s >>= 1) { if (c < s) red[c] += red[c + s]; __syncthreads(); }
  float mean = red[0] * (1.f / CC);
  __syncthreads();
  float d = (c < CC) ? (v - mean) : 0.f;
  red[c] = d * d; __syncthreads();
  for (int s = 128; s > 0; s >>= 1) { if (c < s) red[c] += red[c + s]; __syncthreads(); }
  float inv = rsqrtf(red[0] * (1.f / CC) + 1e-5f);
  if (c < CC) {
    float o = d * inv * lng[c] + lnb[c];
    gated[t * CC + c] = tobf(o * fbf(zs[t * CC + c]));
  }
}

// ================= builders / converters =================
__global__ void k_f2bf(const float* __restrict__ in, bf16_t* __restrict__ out, int n)
{
  int i = blockIdx.x * 256 + threadIdx.x;
  if (i < n) out[i] = tobf(in[i]);
}
__global__ void k_zero(float* __restrict__ p, size_t n)
{
  size_t i = (size_t)blockIdx.x * 256 + threadIdx.x;
  if (i < n) p[i] = 0.f;
}
__global__ void k_build_dct(bf16_t* __restrict__ Wc, bf16_t* __restrict__ WcT)
{
  int i = blockIdx.x * 256 + threadIdx.x;
  if (i >= 64 * 64) return;
  int n = i >> 6, h = i & 63;
  float v = cosf((float)n * ((h + 0.5f) / 64.f) * (float)M_PI) * sqrtf(2.f / 64.f);
  if (n == 0) v *= 0.70710678f;
  Wc[n * 64 + h] = tobf(v);
  WcT[h * 64 + n] = tobf(v);
}
__global__ void k_build_dft(bf16_t* __restrict__ Cc, bf16_t* __restrict__ Sp,
                            bf16_t* __restrict__ Sn, int N)
{
  int i = blockIdx.x * 256 + threadIdx.x;
  if (i >= N * N) return;
  int n = i / N, k = i % N;
  float th = 2.f * (float)M_PI * (float)((n * k) % N) / (float)N;
  Cc[i] = tobf(cosf(th));
  Sp[i] = tobf(sinf(th));
  Sn[i] = tobf(-sinf(th));
}

// ======================================================================
extern "C" void kernel_launch(void* const* d_in, const int* in_sizes, int n_in,
                              void* d_out, int out_size, void* d_ws, size_t ws_size,
                              hipStream_t stream)
{
  (void)in_sizes; (void)n_in; (void)out_size; (void)ws_size;
  const float* x    = (const float*)d_in[0];
  const float* fe   = (const float*)d_in[1];
  const float* gum  = (const float*)d_in[2];
  const float* dww  = (const float*)d_in[3];
  const float* dwb  = (const float*)d_in[4];
  const float* linw = (const float*)d_in[5];
  const float* linb = (const float*)d_in[6];
  const float* p1w  = (const float*)d_in[7];
  const float* p1b  = (const float*)d_in[8];
  const float* g1   = (const float*)d_in[9];
  const float* b1   = (const float*)d_in[10];
  const float* p2w  = (const float*)d_in[11];
  const float* p2b  = (const float*)d_in[12];
  const float* g2   = (const float*)d_in[13];
  const float* b2   = (const float*)d_in[14];
  const float* p3w  = (const float*)d_in[15];
  const float* p3b  = (const float*)d_in[16];
  const float* tokw = (const float*)d_in[17];
  const float* tokb = (const float*)d_in[18];
  const float* lng  = (const float*)d_in[19];
  const float* lnb  = (const float*)d_in[20];
  const float* outw = (const float*)d_in[21];
  const float* outb = (const float*)d_in[22];
  float* out = (float*)d_out;

  char* base = (char*)d_ws;
  size_t off = 0;
  auto alloc = [&](size_t bytes) -> void* {
    void* p = base + off;
    off += (bytes + 255) & ~(size_t)255;
    return p;
  };
  const size_t PLANE = (size_t)TOKN * CC * sizeof(bf16_t);
  bf16_t* P0 = (bf16_t*)alloc(PLANE);
  bf16_t* P1 = (bf16_t*)alloc(PLANE);
  bf16_t* P2 = (bf16_t*)alloc(PLANE);
  bf16_t* P3 = (bf16_t*)alloc(PLANE);
  bf16_t* xh = (bf16_t*)alloc(PLANE);
  bf16_t* zs = (bf16_t*)alloc(PLANE);
  float*  comb   = (float*)alloc((size_t)TOKN * CC * 4);
  float*  kfull  = (float*)alloc((size_t)HWX * CC * 4);
  float*  pooled = (float*)alloc((size_t)BB * CC * 4);
  float*  moe    = (float*)alloc((size_t)BB * 4 * 4);
  bf16_t* linw_b = (bf16_t*)alloc((size_t)C2 * CC * 2);
  bf16_t* outw_b = (bf16_t*)alloc((size_t)CC * CC * 2);
  bf16_t* tokw_b = (bf16_t*)alloc((size_t)CC * CC * 2);
  bf16_t* fe_b   = (bf16_t*)alloc((size_t)HWX * CC * 2);
  bf16_t* Wc     = (bf16_t*)alloc(64 * 64 * 2);
  bf16_t* WcT    = (bf16_t*)alloc(64 * 64 * 2);
  bf16_t* C64c   = (bf16_t*)alloc(64 * 64 * 2);
  bf16_t* S64p   = (bf16_t*)alloc(64 * 64 * 2);
  bf16_t* S64n   = (bf16_t*)alloc(64 * 64 * 2);
  bf16_t* C192c  = (bf16_t*)alloc((size_t)CC * CC * 2);
  bf16_t* S192p  = (bf16_t*)alloc((size_t)CC * CC * 2);
  bf16_t* S192n  = (bf16_t*)alloc((size_t)CC * CC * 2);

  const size_t NELT = (size_t)TOKN * CC;               // 25,165,824
  const int EBLK = (int)((NELT + 255) / 256);          // 98304

  // ---- prep: weights to bf16, transform tables, zero comb ----
  k_f2bf<<<(C2 * CC + 255) / 256, 256, 0, stream>>>(linw, linw_b, C2 * CC);
  k_f2bf<<<(CC * CC + 255) / 256, 256, 0, stream>>>(outw, outw_b, CC * CC);
  k_f2bf<<<(CC * CC + 255) / 256, 256, 0, stream>>>(tokw, tokw_b, CC * CC);
  k_f2bf<<<(HWX * CC + 255) / 256, 256, 0, stream>>>(fe, fe_b, HWX * CC);
  k_build_dct<<<16, 256, 0, stream>>>(Wc, WcT);
  k_build_dft<<<16, 256, 0, stream>>>(C64c, S64p, S64n, 64);
  k_build_dft<<<(CC * CC + 255) / 256, 256, 0, stream>>>(C192c, S192p, S192n, CC);
  k_zero<<<EBLK, 256, 0, stream>>>(comb, NELT);

  // ---- 1) depthwise conv -> tokens ----
  k_dwconv<<<EBLK, 256, 0, stream>>>(x, dww, dwb, P0);

  // ---- 2) token linear 192->384, split xh / silu(z) ----
  k_tok_gemm<<<dim3(TOKN / 64, C2 / 64), 256, 0, stream>>>(
      P0, linw_b, linb, 0, xh, zs, nullptr);

  // ---- 3) router: pool -> MLP -> one-hot moe ----
  k_pool<<<BB, CC, 0, stream>>>(xh, pooled);
  k_mlp<<<1, 256, 0, stream>>>(pooled, p1w, p1b, g1, b1, p2w, p2b, g2, b2,
                               p3w, p3b, gum, moe);

  // ---- 4) k = dec ** relu(freq_embed @ tok_w^T + tok_b) ----
  k_tok_gemm<<<dim3(HWX / 64, CC / 64), 256, 0, stream>>>(
      fe_b, tokw_b, tokb, 3, nullptr, nullptr, kfull);

  // ---- 5) DCT branch ----
  k_caxis<<<dim3(192, 1, BB), 256, 0, stream>>>(Wc, nullptr, nullptr, xh, nullptr,
      P0, nullptr, nullptr, nullptr, 0, 0.f, 1, WWX * CC, 0);
  k_caxis<<<dim3(3, 1, BB * 64), 256, 0, stream>>>(Wc, nullptr, nullptr, P0, nullptr,
      P2, nullptr, nullptr, nullptr, 0, 0.f, 64, CC, 0);
  k_kmul<<<EBLK, 256, 0, stream>>>(P2, nullptr, kfull);
  k_caxis<<<dim3(192, 1, BB), 256, 0, stream>>>(WcT, nullptr, nullptr, P2, nullptr,
      P0, nullptr, nullptr, nullptr, 0, 0.f, 1, WWX * CC, 0);
  k_caxis<<<dim3(3, 1, BB * 64), 256, 0, stream>>>(WcT, nullptr, nullptr, P0, nullptr,
      nullptr, nullptr, comb, moe, 0, 1.f, 64, CC, 1);

  // ---- 6) FFT branch (separable DFT-by-GEMM over H, W, C; batch axis cancels) ----
  k_caxis<<<dim3(192, 1, BB), 256, 0, stream>>>(C64c, S64p, S64n, xh, nullptr,
      P0, P1, nullptr, nullptr, 0, 0.f, 1, WWX * CC, 0);
  k_caxis<<<dim3(3, 1, BB * 64), 256, 0, stream>>>(C64c, S64p, S64n, P0, P1,
      P2, P3, nullptr, nullptr, 0, 0.f, 64, CC, 0);
  k_cchan<<<dim3(TOKN / 64, CC / 64), 256, 0, stream>>>(P2, P3, C192c, S192p, S192n,
      P0, P1, nullptr, nullptr, 0, 0.f, 0);
  k_kmul<<<EBLK, 256, 0, stream>>>(P0, P1, kfull);
  k_cchan<<<dim3(TOKN / 64, CC / 64), 256, 0, stream>>>(P0, P1, C192c, S192n, S192p,
      P2, P3, nullptr, nullptr, 0, 0.f, 0);
  k_caxis<<<dim3(3, 1, BB * 64), 256, 0, stream>>>(C64c, S64n, S64p, P2, P3,
      P0, P1, nullptr, nullptr, 0, 0.f, 64, CC, 0);
  k_caxis<<<dim3(192, 1, BB), 256, 0, stream>>>(C64c, S64n, S64p, P0, P1,
      nullptr, nullptr, comb, moe, 1, 1.f / (64.f * 64.f * 192.f), 1, WWX * CC, 1);

  // ---- 7) Haar branch (per-token, reference's reshape makes it channel-local) ----
  k_haar<<<TOKN, CC, 0, stream>>>(xh, kfull, moe, comb);

  // ---- 8) layernorm + silu(z) gate, final 192->192 GEMM, NCHW transpose ----
  k_ln_gate<<<TOKN, 256, 0, stream>>>(comb, zs, lng, lnb, P1);
  k_tok_gemm<<<dim3(TOKN / 64, CC / 64), 256, 0, stream>>>(
      P1, outw_b, outb, 2, nullptr, nullptr, out);
}
